// SingleViewGATv2_63093069578888
// MI455X (gfx1250) — matrix operand order repaired and checked by reading, not compiled, not run on hardware
//
#include <hip/hip_runtime.h>
#include <hip/hip_bf16.h>
#include <math.h>

typedef __attribute__((ext_vector_type(16))) __bf16 v16bf;
typedef __attribute__((ext_vector_type(8)))  float  v8f;

// ---------------------------------------------------------------- helpers
__device__ __forceinline__ unsigned short f32_to_bf16_bits(float f) {
  unsigned int u = __float_as_uint(f);
  unsigned int r = ((u >> 16) & 1u) + 0x7FFFu;   // round-to-nearest-even
  return (unsigned short)((u + r) >> 16);
}

__device__ __forceinline__ void atomicMaxF32(float* addr, float val) {
  // monotonic float<->int trick; location initialized to -inf
  if (val >= 0.0f) atomicMax((int*)addr, __float_as_int(val));
  else             atomicMin((unsigned int*)addr, __float_as_uint(val));
}

__device__ __forceinline__ float leaky02(float v) { return v > 0.0f ? v : 0.2f * v; }

// ---------------------------------------------------------------- fills / casts
__global__ void fill_f32_kernel(float* __restrict__ p, float v, size_t n) {
  size_t i = (size_t)blockIdx.x * blockDim.x + threadIdx.x;
  if (i < n) p[i] = v;
}

__global__ void cast_f32_bf16_kernel(const float* __restrict__ src,
                                     unsigned short* __restrict__ dst, size_t n) {
  size_t i = (size_t)blockIdx.x * blockDim.x + threadIdx.x;
  if (i < n) dst[i] = f32_to_bf16_bits(src[i]);
}

// W[K][C] f32 row-major  ->  Wt[C][K] bf16 (K contiguous, so K-pairs are dwords)
__global__ void cast_transpose_bf16_kernel(const float* __restrict__ W,
                                           unsigned short* __restrict__ Wt,
                                           int K, int C) {
  int idx = blockIdx.x * blockDim.x + threadIdx.x;
  if (idx >= K * C) return;
  int c = idx / K, k = idx % K;
  Wt[(size_t)c * K + k] = f32_to_bf16_bits(W[(size_t)k * C + c]);
}

// ---------------------------------------------------------------- WMMA GEMM
// C[nrows][ncols] = Xb[nrows][K](bf16) * Wt[ncols][K](bf16, transposed W) + bias
// block = 256 threads = 8 waves; block tile = 64 rows x 128 cols
// wave tile = 64 rows x 16 cols: 4 accumulators, B fragment loaded once per
// K-step and reused across 4 WMMAs (4x arithmetic intensity on the B stream).
__global__ void __launch_bounds__(256)
gemm_bf16_wmma_kernel(const unsigned int* __restrict__ Xb32,   // [nrows][K/2] packed bf16 pairs
                      const unsigned int* __restrict__ Wt32,   // [ncols][K/2] packed bf16 pairs
                      const float* __restrict__ bias,
                      float* __restrict__ C,
                      int nrows, int K, int ncols) {
  extern __shared__ unsigned int lA[];            // 64 rows x (K/2) dwords
  const int tid  = threadIdx.x;
  const int lane = tid & 31;
  const int wave = tid >> 5;
  const int row0 = blockIdx.x * 64;
  const int c0   = blockIdx.y * 128 + wave * 16;
  const int Kw   = K >> 1;                        // dwords per row

  const int m  = lane & 15;      // A/D row-in-tile, B/D column index
  const int hl = lane >> 4;      // lane half-group

  // warm L2 with this wave's B panel while the A tile is being staged
  if (c0 < ncols) __builtin_prefetch(Wt32 + (size_t)(c0 + m) * Kw, 0, 1);

  // cooperative A-tile load: global -> LDS (zero-pad past nrows)
  for (int w = tid; w < 64 * Kw; w += 256) {
    int r = w / Kw, kw = w - r * Kw;
    unsigned int val = 0u;
    int gr = row0 + r;
    if (gr < nrows) val = Xb32[(size_t)gr * Kw + kw];
    lA[r * Kw + kw] = val;
  }
  __syncthreads();

  if (c0 >= ncols) return;

  union { unsigned int u[8]; v16bf v; } a, b;
  v8f acc[4] = {};

  for (int kk = 0; kk < K; kk += 32) {
    // B fragment (32x16 bf16): VGPR q -> K = kk + hl*16 + 2q  (contiguous dwords)
#pragma unroll
    for (int q = 0; q < 8; ++q) {
      int kb = kk + hl * 16 + 2 * q;
      b.u[q] = Wt32[(size_t)(c0 + m) * Kw + (kb >> 1)];
    }
#pragma unroll
    for (int mt = 0; mt < 4; ++mt) {
      // A fragment (16x32 bf16): VGPR q -> K = kk + hl*16 + 2q  (contiguous dwords)
#pragma unroll
      for (int q = 0; q < 8; ++q) {
        int ka = kk + hl * 16 + 2 * q;
        a.u[q] = lA[(mt * 16 + m) * Kw + (ka >> 1)];
      }
      acc[mt] = __builtin_amdgcn_wmma_f32_16x16x32_bf16(false, a.v, false, b.v,
                                                        (short)0, acc[mt], false, false);
    }
  }

  const float bc = bias[c0 + m];
#pragma unroll
  for (int mt = 0; mt < 4; ++mt) {
#pragma unroll
    for (int q = 0; q < 8; ++q) {
      int gr = row0 + mt * 16 + q + 8 * hl;      // D layout: M = q + 8*half
      if (gr < nrows) C[(size_t)gr * ncols + c0 + m] = acc[mt][q] + bc;
    }
  }
}

// ---------------------------------------------------------------- layer-1 edge ops (H=8, C=64)
__global__ void edge_score1_kernel(const int* __restrict__ ei, int E, int ET,
                                   const float* __restrict__ XL, const float* __restrict__ XR,
                                   const float* __restrict__ att,
                                   float* __restrict__ Esc, float* __restrict__ Mx) {
  int idx = blockIdx.x * blockDim.x + threadIdx.x;
  if (idx >= ET * 8) return;
  int e = idx >> 3, h = idx & 7;
  int s, d;
  if (e < E) { s = ei[e]; d = ei[E + e]; } else { s = d = e - E; }
  const float* xl = XL + (size_t)s * 512 + h * 64;
  const float* xr = XR + (size_t)d * 512 + h * 64;
  const float* at = att + h * 64;
  float acc = 0.0f;
#pragma unroll
  for (int c = 0; c < 64; c += 4) {
    float4 av = *(const float4*)(xl + c);
    float4 bv = *(const float4*)(xr + c);
    float4 wv = *(const float4*)(at + c);
    acc += leaky02(av.x + bv.x) * wv.x;
    acc += leaky02(av.y + bv.y) * wv.y;
    acc += leaky02(av.z + bv.z) * wv.z;
    acc += leaky02(av.w + bv.w) * wv.w;
  }
  Esc[idx] = acc;
  atomicMaxF32(&Mx[(size_t)d * 8 + h], acc);
}

__global__ void edge_p1_kernel(const int* __restrict__ ei, int E, int ET,
                               float* __restrict__ Esc, const float* __restrict__ Mx,
                               float* __restrict__ S) {
  int idx = blockIdx.x * blockDim.x + threadIdx.x;
  if (idx >= ET * 8) return;
  int e = idx >> 3, h = idx & 7;
  int d = (e < E) ? ei[E + e] : (e - E);
  float p = expf(Esc[idx] - Mx[(size_t)d * 8 + h]);
  Esc[idx] = p;
  atomicAdd(&S[(size_t)d * 8 + h], p);
}

// one block per edge: 128 threads = 8 heads x 16 groups of 4 channels
__global__ void edge_agg1_kernel(const int* __restrict__ ei, int E, int ET,
                                 const float* __restrict__ XL, const float* __restrict__ P,
                                 const float* __restrict__ S, float* __restrict__ ACC) {
  int e = blockIdx.x;
  int t = threadIdx.x;
  int h = t >> 4, c = (t & 15) * 4;
  int s, d;
  if (e < E) { s = ei[e]; d = ei[E + e]; } else { s = d = e - E; }
  float alpha = P[(size_t)e * 8 + h] / S[(size_t)d * 8 + h];
  float4 xv = *(const float4*)(XL + (size_t)s * 512 + h * 64 + c);
  float* out = ACC + (size_t)d * 512 + h * 64 + c;
  atomicAdd(out + 0, alpha * xv.x);
  atomicAdd(out + 1, alpha * xv.y);
  atomicAdd(out + 2, alpha * xv.z);
  atomicAdd(out + 3, alpha * xv.w);
}

__global__ void bias_relu_bf16_kernel(const float* __restrict__ ACC,
                                      const float* __restrict__ bias,
                                      unsigned short* __restrict__ H, size_t total, int C) {
  size_t i = (size_t)blockIdx.x * blockDim.x + threadIdx.x;
  if (i >= total) return;
  int c = (int)(i % C);
  float v = ACC[i] + bias[c];
  H[i] = f32_to_bf16_bits(v > 0.0f ? v : 0.0f);
}

// ---------------------------------------------------------------- layer-2 edge ops (H=1, C=64)
__global__ void edge_score2_kernel(const int* __restrict__ ei, int E, int ET,
                                   const float* __restrict__ XL, const float* __restrict__ XR,
                                   const float* __restrict__ att,
                                   float* __restrict__ Esc, float* __restrict__ Mx) {
  int e = blockIdx.x * blockDim.x + threadIdx.x;
  if (e >= ET) return;
  int s, d;
  if (e < E) { s = ei[e]; d = ei[E + e]; } else { s = d = e - E; }
  const float* xl = XL + (size_t)s * 64;
  const float* xr = XR + (size_t)d * 64;
  float acc = 0.0f;
#pragma unroll
  for (int c = 0; c < 64; c += 4) {
    float4 av = *(const float4*)(xl + c);
    float4 bv = *(const float4*)(xr + c);
    float4 wv = *(const float4*)(att + c);
    acc += leaky02(av.x + bv.x) * wv.x;
    acc += leaky02(av.y + bv.y) * wv.y;
    acc += leaky02(av.z + bv.z) * wv.z;
    acc += leaky02(av.w + bv.w) * wv.w;
  }
  Esc[e] = acc;
  atomicMaxF32(&Mx[d], acc);
}

__global__ void edge_p2_kernel(const int* __restrict__ ei, int E, int ET,
                               float* __restrict__ Esc, const float* __restrict__ Mx,
                               float* __restrict__ S) {
  int e = blockIdx.x * blockDim.x + threadIdx.x;
  if (e >= ET) return;
  int d = (e < E) ? ei[E + e] : (e - E);
  float p = expf(Esc[e] - Mx[d]);
  Esc[e] = p;
  atomicAdd(&S[d], p);
}

__global__ void edge_agg2_kernel(const int* __restrict__ ei, int E, int ET,
                                 const float* __restrict__ XL, const float* __restrict__ P,
                                 const float* __restrict__ S, float* __restrict__ ACC) {
  int e = blockIdx.x;
  int c = threadIdx.x;          // 64 channels
  int s, d;
  if (e < E) { s = ei[e]; d = ei[E + e]; } else { s = d = e - E; }
  float alpha = P[e] / S[d];
  atomicAdd(&ACC[(size_t)d * 64 + c], alpha * XL[(size_t)s * 64 + c]);
}

// ---------------------------------------------------------------- pooling + head
__global__ void pool_kernel(const float* __restrict__ ACC2, const float* __restrict__ bias2,
                            const int* __restrict__ batch,
                            float* __restrict__ POOL, float* __restrict__ CNT, int N) {
  int idx = blockIdx.x * blockDim.x + threadIdx.x;
  if (idx >= N * 64) return;
  int n = idx >> 6, c = idx & 63;
  float v = ACC2[(size_t)n * 64 + c] + bias2[c];
  int g = batch[n];
  atomicAdd(&POOL[(size_t)g * 64 + c], v);
  if (c == 0) atomicAdd(&CNT[g], 1.0f);
}

__global__ void head_kernel(const float* __restrict__ POOL, const float* __restrict__ CNT,
                            const float* __restrict__ GF,
                            const float* __restrict__ W1, const float* __restrict__ B1,
                            const float* __restrict__ W2, const float* __restrict__ B2,
                            float* __restrict__ out) {
  __shared__ float in[80];
  __shared__ float z[64];
  int g = blockIdx.x, j = threadIdx.x;   // 96 threads
  float cnt = fmaxf(CNT[g], 1.0f);
  if (j < 64)      in[j] = POOL[(size_t)g * 64 + j] / cnt;
  else if (j < 80) in[j] = GF[(size_t)g * 16 + (j - 64)];
  __syncthreads();
  if (j < 64) {
    float s = B1[j];
    for (int k = 0; k < 80; ++k) s += in[k] * W1[k * 64 + j];
    z[j] = s > 0.0f ? s : 0.0f;
  }
  __syncthreads();
  if (j < 10) {
    float s = B2[j];
    for (int i = 0; i < 64; ++i) s += z[i] * W2[i * 10 + j];
    out[(size_t)g * 10 + j] = s;
  }
}

// ---------------------------------------------------------------- launch
extern "C" void kernel_launch(void* const* d_in, const int* in_sizes, int n_in,
                              void* d_out, int out_size, void* d_ws, size_t ws_size,
                              hipStream_t stream) {
  const float* x     = (const float*)d_in[0];
  const int*   ei    = (const int*)d_in[1];
  const int*   batch = (const int*)d_in[2];
  const float* gf    = (const float*)d_in[3];
  const float* W1l   = (const float*)d_in[4];
  const float* b1l   = (const float*)d_in[5];
  const float* W1r   = (const float*)d_in[6];
  const float* b1r   = (const float*)d_in[7];
  const float* att1  = (const float*)d_in[8];
  const float* bias1 = (const float*)d_in[9];
  const float* W2l   = (const float*)d_in[10];
  const float* b2l   = (const float*)d_in[11];
  const float* W2r   = (const float*)d_in[12];
  const float* b2r   = (const float*)d_in[13];
  const float* att2  = (const float*)d_in[14];
  const float* bias2 = (const float*)d_in[15];
  const float* l1W   = (const float*)d_in[16];
  const float* l1b   = (const float*)d_in[17];
  const float* l2W   = (const float*)d_in[18];
  const float* l2b   = (const float*)d_in[19];

  const int Nn = in_sizes[0] / 64;       // 20000
  const int Ee = in_sizes[1] / 2;        // 320000
  const int ET = Ee + Nn;                // edges + self loops
  const int Gg = in_sizes[3] / 16;       // 64

  // workspace carve-out
  size_t off = 0;
  auto alloc = [&](size_t bytes) -> char* {
    char* p = (char*)d_ws + off;
    off += (bytes + 255) & ~(size_t)255;
    return p;
  };
  unsigned short* XB   = (unsigned short*)alloc((size_t)Nn * 64 * 2);
  unsigned short* W1LT = (unsigned short*)alloc((size_t)512 * 64 * 2);
  unsigned short* W1RT = (unsigned short*)alloc((size_t)512 * 64 * 2);
  unsigned short* W2LT = (unsigned short*)alloc((size_t)64 * 512 * 2);
  unsigned short* W2RT = (unsigned short*)alloc((size_t)64 * 512 * 2);
  float* XL1  = (float*)alloc((size_t)Nn * 512 * 4);
  float* XR1  = (float*)alloc((size_t)Nn * 512 * 4);
  float* ACC1 = (float*)alloc((size_t)Nn * 512 * 4);
  float* E1   = (float*)alloc((size_t)ET * 8 * 4);
  float* M1   = (float*)alloc((size_t)Nn * 8 * 4);
  float* S1   = (float*)alloc((size_t)Nn * 8 * 4);
  unsigned short* H1B = (unsigned short*)alloc((size_t)Nn * 512 * 2);
  float* XL2  = (float*)alloc((size_t)Nn * 64 * 4);
  float* XR2  = (float*)alloc((size_t)Nn * 64 * 4);
  float* ACC2 = (float*)alloc((size_t)Nn * 64 * 4);
  float* E2   = (float*)alloc((size_t)ET * 4);
  float* M2   = (float*)alloc((size_t)Nn * 4);
  float* S2   = (float*)alloc((size_t)Nn * 4);
  float* POOL = (float*)alloc((size_t)Gg * 64 * 4);
  float* CNT  = (float*)alloc((size_t)Gg * 4);
  (void)ws_size; (void)n_in; (void)out_size;

  const int T = 256;
  auto nblk = [](size_t n, int t) { return (unsigned)((n + t - 1) / t); };

  // casts
  cast_f32_bf16_kernel<<<nblk((size_t)Nn * 64, T), T, 0, stream>>>(x, XB, (size_t)Nn * 64);
  cast_transpose_bf16_kernel<<<nblk(512 * 64, T), T, 0, stream>>>(W1l, W1LT, 64, 512);
  cast_transpose_bf16_kernel<<<nblk(512 * 64, T), T, 0, stream>>>(W1r, W1RT, 64, 512);
  cast_transpose_bf16_kernel<<<nblk(512 * 64, T), T, 0, stream>>>(W2l, W2LT, 512, 64);
  cast_transpose_bf16_kernel<<<nblk(512 * 64, T), T, 0, stream>>>(W2r, W2RT, 512, 64);

  // accumulator init (every call: deterministic)
  fill_f32_kernel<<<nblk((size_t)Nn * 8, T), T, 0, stream>>>(M1, -INFINITY, (size_t)Nn * 8);
  fill_f32_kernel<<<nblk((size_t)Nn * 8, T), T, 0, stream>>>(S1, 0.0f, (size_t)Nn * 8);
  fill_f32_kernel<<<nblk((size_t)Nn * 512, T), T, 0, stream>>>(ACC1, 0.0f, (size_t)Nn * 512);
  fill_f32_kernel<<<nblk((size_t)Nn, T), T, 0, stream>>>(M2, -INFINITY, (size_t)Nn);
  fill_f32_kernel<<<nblk((size_t)Nn, T), T, 0, stream>>>(S2, 0.0f, (size_t)Nn);
  fill_f32_kernel<<<nblk((size_t)Nn * 64, T), T, 0, stream>>>(ACC2, 0.0f, (size_t)Nn * 64);
  fill_f32_kernel<<<nblk((size_t)Gg * 64, T), T, 0, stream>>>(POOL, 0.0f, (size_t)Gg * 64);
  fill_f32_kernel<<<nblk((size_t)Gg, T), T, 0, stream>>>(CNT, 0.0f, (size_t)Gg);

  // layer 1 GEMMs: [Nn,64] x [64,512]; A tile LDS = 64 rows * 64 bf16 = 8KB
  dim3 g1((Nn + 63) / 64, 512 / 128);
  size_t lds1 = (size_t)64 * (64 / 2) * 4;
  gemm_bf16_wmma_kernel<<<g1, 256, lds1, stream>>>((const unsigned int*)XB,
                                                   (const unsigned int*)W1LT,
                                                   b1l, XL1, Nn, 64, 512);
  gemm_bf16_wmma_kernel<<<g1, 256, lds1, stream>>>((const unsigned int*)XB,
                                                   (const unsigned int*)W1RT,
                                                   b1r, XR1, Nn, 64, 512);

  // layer 1 attention
  edge_score1_kernel<<<nblk((size_t)ET * 8, T), T, 0, stream>>>(ei, Ee, ET, XL1, XR1, att1, E1, M1);
  edge_p1_kernel<<<nblk((size_t)ET * 8, T), T, 0, stream>>>(ei, Ee, ET, E1, M1, S1);
  edge_agg1_kernel<<<(unsigned)ET, 128, 0, stream>>>(ei, Ee, ET, XL1, E1, S1, ACC1);
  bias_relu_bf16_kernel<<<nblk((size_t)Nn * 512, T), T, 0, stream>>>(ACC1, bias1, H1B,
                                                                     (size_t)Nn * 512, 512);

  // layer 2 GEMMs: [Nn,512] x [512,64]; A tile LDS = 64 rows * 512 bf16 = 64KB
  dim3 g2((Nn + 63) / 64, 1);
  size_t lds2 = (size_t)64 * (512 / 2) * 4;
  gemm_bf16_wmma_kernel<<<g2, 256, lds2, stream>>>((const unsigned int*)H1B,
                                                   (const unsigned int*)W2LT,
                                                   b2l, XL2, Nn, 512, 64);
  gemm_bf16_wmma_kernel<<<g2, 256, lds2, stream>>>((const unsigned int*)H1B,
                                                   (const unsigned int*)W2RT,
                                                   b2r, XR2, Nn, 512, 64);

  // layer 2 attention
  edge_score2_kernel<<<nblk((size_t)ET, T), T, 0, stream>>>(ei, Ee, ET, XL2, XR2, att2, E2, M2);
  edge_p2_kernel<<<nblk((size_t)ET, T), T, 0, stream>>>(ei, Ee, ET, E2, M2, S2);
  edge_agg2_kernel<<<(unsigned)ET, 64, 0, stream>>>(ei, Ee, ET, XL2, E2, S2, ACC2);

  // pooling + MLP head
  pool_kernel<<<nblk((size_t)Nn * 64, T), T, 0, stream>>>(ACC2, bias2, batch, POOL, CNT, Nn);
  head_kernel<<<(unsigned)Gg, 96, 0, stream>>>(POOL, CNT, gf, l1W, l1b, l2W, l2b, (float*)d_out);
}
